// MultiAttention_27668179321031
// MI455X (gfx1250) — compile-verified, matrix operand order checked by reading, not run on hardware
//
#include <hip/hip_runtime.h>
#include <hip/hip_bf16.h>

// ---------------------------------------------------------------------------
// MultiAttention forward for MI455X (gfx1250, wave32, WMMA bf16 path)
// out  = ((q@Wq+bq) attn (k@Wk+bk, v@Wv+bv)) @ Wout + bout      (L,B,DIM) f32
// out2 = attn.mean(over heads)                                   (B,L,N)  f32
// ---------------------------------------------------------------------------

typedef __attribute__((ext_vector_type(16))) __bf16 v16bf;
typedef __attribute__((ext_vector_type(8)))  __bf16 v8bf;
typedef __attribute__((ext_vector_type(8)))  float  v8f;
typedef int vint4 __attribute__((vector_size(16)));   // matches builtin param

#define DEV __device__ __forceinline__

constexpr int Lq  = 1024;   // query length
constexpr int Nk  = 1024;   // key length
constexpr int Bb  = 8;      // batch
constexpr int DIM = 1024;   // model dim
constexpr int NH  = 16;     // heads

// ---- gfx1250 async global->LDS copy (ASYNCcnt path), with safe fallback ----
#if defined(__has_builtin)
#if __has_builtin(__builtin_amdgcn_global_load_async_to_lds_b128) && \
    __has_builtin(__builtin_amdgcn_s_wait_asynccnt)
#define USE_ASYNC_LDS 1
#endif
#endif
#ifndef USE_ASYNC_LDS
#define USE_ASYNC_LDS 0
#endif

DEV void copy16_to_lds(__bf16* lds, const __bf16* gsrc) {
#if USE_ASYNC_LDS
  __builtin_amdgcn_global_load_async_to_lds_b128(
      (__attribute__((address_space(1))) vint4*)gsrc,
      (__attribute__((address_space(3))) vint4*)lds, 0, 0);
#else
  *reinterpret_cast<v8bf*>(lds) = *reinterpret_cast<const v8bf*>(gsrc);
#endif
}

DEV void copy_fence() {
#if USE_ASYNC_LDS
  __builtin_amdgcn_s_wait_asynccnt(0);
#endif
}

// 16-bit A/B fragment (16x32 MxK pattern) from a bf16 row: lanes 0-15 hold
// K{0..7,16..23}, lanes 16-31 hold K{8..15,24..31} (ISA 7.12.2).
DEV v16bf frag_b16(const __bf16* row, int lane) {
  const __bf16* p = row + ((lane & 16) ? 8 : 0);
  v8bf lo = *reinterpret_cast<const v8bf*>(p);        // 16B
  v8bf hi = *reinterpret_cast<const v8bf*>(p + 16);   // 16B
  return __builtin_shufflevector(lo, hi, 0,1,2,3,4,5,6,7,8,9,10,11,12,13,14,15);
}

// Same fragment built from an f32 row (attention probabilities in LDS).
DEV v16bf frag_f32(const float* row, int lane) {
  const int kb = (lane & 16) ? 8 : 0;
  v16bf r;
#pragma unroll
  for (int i = 0; i < 8; ++i) {
    r[i]     = (__bf16)row[kb + i];
    r[i + 8] = (__bf16)row[kb + 16 + i];
  }
  return r;
}

DEV v8f wmma_bf16(v16bf a, v16bf b, v8f c) {
  return __builtin_amdgcn_wmma_f32_16x16x32_bf16(false, a, false, b,
                                                 (short)0, c, false, false);
}

DEV float red_max8(float v) {
  v = fmaxf(v, __shfl_xor(v, 1));
  v = fmaxf(v, __shfl_xor(v, 2));
  v = fmaxf(v, __shfl_xor(v, 4));
  return v;
}
DEV float red_sum8(float v) {
  v += __shfl_xor(v, 1);
  v += __shfl_xor(v, 2);
  v += __shfl_xor(v, 4);
  return v;
}

// ---------------------------------------------------------------------------
// Pre-pass 1: elementwise f32 -> bf16 (8 elems / thread)
// ---------------------------------------------------------------------------
__global__ __launch_bounds__(256) void cvt_bf16(const float* __restrict__ in,
                                                __bf16* __restrict__ out, int n8) {
  const int i = blockIdx.x * 256 + threadIdx.x;
  if (i >= n8) return;
  const float4* p = reinterpret_cast<const float4*>(in) + (size_t)i * 2;
  const float4 a = p[0], b = p[1];
  v8bf o;
  o[0] = (__bf16)a.x; o[1] = (__bf16)a.y; o[2] = (__bf16)a.z; o[3] = (__bf16)a.w;
  o[4] = (__bf16)b.x; o[5] = (__bf16)b.y; o[6] = (__bf16)b.z; o[7] = (__bf16)b.w;
  reinterpret_cast<v8bf*>(out)[i] = o;
}

// ---------------------------------------------------------------------------
// Pre-pass 2: weight convert + transpose: W (K,N) f32 -> Wt (N,K) bf16
// ---------------------------------------------------------------------------
__global__ __launch_bounds__(256) void transpose_cvt(const float* __restrict__ W,
                                                     __bf16* __restrict__ Wt) {
  __shared__ float tile[32][33];
  const int tx = threadIdx.x & 31;
  const int ty = threadIdx.x >> 5;          // 0..7
  const int k0 = blockIdx.x * 32;
  const int n0 = blockIdx.y * 32;
#pragma unroll
  for (int i = 0; i < 4; ++i)
    tile[ty + i * 8][tx] = W[(size_t)(k0 + ty + i * 8) * DIM + n0 + tx];
  __syncthreads();
#pragma unroll
  for (int i = 0; i < 4; ++i)
    Wt[(size_t)(n0 + ty + i * 8) * DIM + k0 + tx] = (__bf16)tile[tx][ty + i * 8];
}

// ---------------------------------------------------------------------------
// GEMM: C[M=8192, N=1024] = A[M,1024] @ Bt[N,1024]^T + bias  (all bf16 in)
// MODE 0/1: out bf16 scattered to Q/K layout (b,h,n,64)
// MODE 2:   out bf16 scattered to V^T layout (b,h,64,n)
// MODE 3:   out f32 row-major (final projection)
// ---------------------------------------------------------------------------
template <int MODE>
__global__ __launch_bounds__(256) void gemm_wmma(
    const __bf16* __restrict__ A, const __bf16* __restrict__ Bt,
    const float* __restrict__ bias, void* __restrict__ Out) {
  constexpr int BM = 128, BN = 128, BK = 32, LDT = 40;  // pad keeps 16B align
  constexpr int K = 1024, NC = 1024;
  __shared__ __bf16 sA[BM * LDT];
  __shared__ __bf16 sB[BN * LDT];

  const int t    = threadIdx.x;
  const int lane = t & 31;
  const int wid  = t >> 5;
  const int m0   = blockIdx.x * BM;
  const int n0   = blockIdx.y * BN;
  const int wm   = (wid >> 2) * 64;
  const int wn   = (wid & 3) * 32;

  const __bf16* Arows = A + (size_t)m0 * K;
  const __bf16* Brows = Bt + (size_t)n0 * K;

  v8f acc[4][2];
#pragma unroll
  for (int mi = 0; mi < 4; ++mi)
#pragma unroll
    for (int ni = 0; ni < 2; ++ni) acc[mi][ni] = {};

  for (int k0 = 0; k0 < K; k0 += BK) {
    // stage both 128x32 bf16 tiles: 512 16B-chunks each, 2 per thread
#pragma unroll
    for (int cc = 0; cc < 2; ++cc) {
      const int c = t + cc * 256;
      const int row = c >> 2, qd = c & 3;
      copy16_to_lds(sA + row * LDT + qd * 8,
                    Arows + (size_t)row * K + k0 + qd * 8);
      copy16_to_lds(sB + row * LDT + qd * 8,
                    Brows + (size_t)row * K + k0 + qd * 8);
    }
    copy_fence();
    __syncthreads();

    v16bf af[4], bfv[2];
#pragma unroll
    for (int mi = 0; mi < 4; ++mi)
      af[mi] = frag_b16(sA + (wm + mi * 16 + (lane & 15)) * LDT, lane);
#pragma unroll
    for (int ni = 0; ni < 2; ++ni)
      bfv[ni] = frag_b16(sB + (wn + ni * 16 + (lane & 15)) * LDT, lane);
#pragma unroll
    for (int mi = 0; mi < 4; ++mi)
#pragma unroll
      for (int ni = 0; ni < 2; ++ni)
        acc[mi][ni] = wmma_bf16(af[mi], bfv[ni], acc[mi][ni]);
    __syncthreads();
  }

  // C tile layout: VGPR g, lane l -> M = g + 8*(l>>4), N = l&15
#pragma unroll
  for (int mi = 0; mi < 4; ++mi)
#pragma unroll
    for (int ni = 0; ni < 2; ++ni) {
      const int mBase = m0 + wm + mi * 16 + ((lane >> 4) << 3);
      const int n     = n0 + wn + ni * 16 + (lane & 15);
      const float bv  = bias[n];
#pragma unroll
      for (int g = 0; g < 8; ++g) {
        const int r = mBase + g;           // global row = seq*B + batch
        const float val = acc[mi][ni][g] + bv;
        if (MODE == 3) {
          reinterpret_cast<float*>(Out)[(size_t)r * NC + n] = val;
        } else {
          const int bi = r & 7;            // B == 8
          const int nn = r >> 3;           // seq index
          const int h  = n >> 6, d = n & 63;
          size_t idx;
          if (MODE == 2)
            idx = (((size_t)bi * NH + h) * 64 + d) * (size_t)Nk + nn;    // V^T
          else
            idx = (((size_t)bi * NH + h) * 1024 + nn) * 64 + d;          // Q/K
          reinterpret_cast<__bf16*>(Out)[idx] = (__bf16)val;
        }
      }
    }
}

// ---------------------------------------------------------------------------
// Attention: one block per (batch, 32-row i-tile), loops all 16 heads.
// LDS: S[32][1028] scores/probs (f32) + MeanAcc[32][1028] head-mean accum.
// ---------------------------------------------------------------------------
__global__ __launch_bounds__(256) void attn_wmma(
    const __bf16* __restrict__ Qp, const __bf16* __restrict__ Kp,
    const __bf16* __restrict__ VT, const float* __restrict__ mask,
    __bf16* __restrict__ CTX, float* __restrict__ meanOut) {
  constexpr int TM = 32, SROW = 1028;     // pad rotates LDS banks
  extern __shared__ float smem[];
  float* S  = smem;                       // TM * SROW
  float* MA = smem + TM * SROW;           // TM * SROW

  const int t    = threadIdx.x;
  const int lane = t & 31;
  const int wid  = t >> 5;
  const int i0   = blockIdx.x * TM;
  const int b    = blockIdx.y;

  for (int idx = t; idx < TM * SROW; idx += 256) MA[idx] = 0.f;
  __syncthreads();

  const int row = t >> 3;   // 0..31 (8 lanes per row -> shuffle reductions)
  const int seg = t & 7;

  for (int h = 0; h < NH; ++h) {
    const __bf16* Qb = Qp + (((size_t)b * NH + h) * Lq + i0) * 64;
    const __bf16* Kb = Kp + ((size_t)b * NH + h) * (size_t)Nk * 64;
    const __bf16* Vb = VT + ((size_t)b * NH + h) * 64 * (size_t)Nk;

    // Q fragments for this head (32 rows x 64 head-dim), per wave
    v16bf qf[2][2];
#pragma unroll
    for (int mi = 0; mi < 2; ++mi)
#pragma unroll
      for (int kc = 0; kc < 2; ++kc)
        qf[mi][kc] = frag_b16(Qb + (mi * 16 + (lane & 15)) * 64 + kc * 32, lane);

    // ---- scores: S = (Q K^T) * scale ----
    for (int jt = wid; jt < Nk / 16; jt += 8) {
      const int j0 = jt * 16;
      const __bf16* Krow = Kb + (j0 + (lane & 15)) * 64;
      v16bf kf0 = frag_b16(Krow, lane);
      v16bf kf1 = frag_b16(Krow + 32, lane);
      v8f c0 = {}, c1 = {};
      c0 = wmma_bf16(qf[0][0], kf0, c0);
      c0 = wmma_bf16(qf[0][1], kf1, c0);
      c1 = wmma_bf16(qf[1][0], kf0, c1);
      c1 = wmma_bf16(qf[1][1], kf1, c1);
      const int n  = j0 + (lane & 15);
      const int mb = (lane >> 4) << 3;
#pragma unroll
      for (int g = 0; g < 8; ++g) {
        S[(mb + g) * SROW + n]      = c0[g] * 0.125f;   // D^-0.5
        S[(16 + mb + g) * SROW + n] = c1[g] * 0.125f;
      }
    }
    __syncthreads();

    // ---- softmax, post-softmax mask, renorm:  p*m / (sum(p*m)+1e-4*sum(p)) ----
    {
      float* Sr = S + row * SROW;
      float mx = -3.0e38f;
      for (int jj = 0; jj < 128; ++jj) mx = fmaxf(mx, Sr[seg + jj * 8]);
      mx = red_max8(mx);
      float s1 = 0.f;
      for (int jj = 0; jj < 128; ++jj) {
        const int j = seg + jj * 8;
        const float p = __expf(Sr[j] - mx);
        Sr[j] = p;
        s1 += p;
      }
      s1 = red_sum8(s1);
      const float* Mr = mask + (((size_t)b * NH + h) * Lq + (i0 + row)) * (size_t)Nk;
      float s2 = 0.f;
      for (int jj = 0; jj < 128; ++jj) {
        const int j = seg + jj * 8;
        const float pm = Sr[j] * Mr[j];
        Sr[j] = pm;
        s2 += pm;
      }
      s2 = red_sum8(s2);
      const float inv = 1.f / (s2 + 1e-4f * s1);
      float* Mar = MA + row * SROW;
      for (int jj = 0; jj < 128; ++jj) {
        const int j = seg + jj * 8;
        const float f = Sr[j] * inv;
        Sr[j] = f;
        Mar[j] += f;
      }
    }
    __syncthreads();

    // ---- context: ctx = attn @ V  (V^T rows are contiguous over j) ----
    {
      const int mt = (wid >> 2) * 16;   // 0/16
      const int dt = (wid & 3) * 16;    // 0..48
      const __bf16* Vrow = Vb + (dt + (lane & 15)) * (size_t)Nk;
      const float* Arow = S + (mt + (lane & 15)) * SROW;
      v8f acc = {};
      for (int jc = 0; jc < Nk; jc += 32) {
        v16bf a  = frag_f32(Arow + jc, lane);
        v16bf bv = frag_b16(Vrow + jc, lane);
        acc = wmma_bf16(a, bv, acc);
      }
      const int mb = (lane >> 4) << 3;
      const int d  = dt + (lane & 15);
#pragma unroll
      for (int g = 0; g < 8; ++g) {
        const int li = i0 + mt + mb + g;
        CTX[((size_t)li * Bb + b) * DIM + h * 64 + d] = (__bf16)acc[g];
      }
    }
    __syncthreads();
  }

  // ---- attn.mean over heads -> (B, L, N) ----
  const float* Mar = MA + row * SROW;
  float* Mo = meanOut + ((size_t)b * Lq + (i0 + row)) * (size_t)Nk;
  for (int jj = 0; jj < 128; ++jj) {
    const int j = seg + jj * 8;
    Mo[j] = Mar[j] * (1.0f / 16.0f);
  }
}

// ---------------------------------------------------------------------------
extern "C" void kernel_launch(void* const* d_in, const int* in_sizes, int n_in,
                              void* d_out, int out_size, void* d_ws, size_t ws_size,
                              hipStream_t stream) {
  (void)in_sizes; (void)n_in; (void)out_size; (void)ws_size;
  const float* q    = (const float*)d_in[0];
  const float* k    = (const float*)d_in[1];
  const float* v    = (const float*)d_in[2];
  const float* mask = (const float*)d_in[3];
  const float* Wq   = (const float*)d_in[4];
  const float* bq   = (const float*)d_in[5];
  const float* Wk   = (const float*)d_in[6];
  const float* bk   = (const float*)d_in[7];
  const float* Wv   = (const float*)d_in[8];
  const float* bv   = (const float*)d_in[9];
  const float* Wout = (const float*)d_in[10];
  const float* bout = (const float*)d_in[11];

  constexpr size_t actB = (size_t)Lq * Bb * DIM * 2;   // 16.78 MB per act buffer
  constexpr size_t wB   = (size_t)DIM * DIM * 2;       // 2.1 MB per weight
  char* ws = (char*)d_ws;
  __bf16* qb    = (__bf16*)(ws);
  __bf16* kb2   = (__bf16*)(ws + actB);
  __bf16* vb2   = (__bf16*)(ws + actB * 2);
  __bf16* Qp    = (__bf16*)(ws + actB * 3);
  __bf16* Kp    = (__bf16*)(ws + actB * 4);
  __bf16* VT    = (__bf16*)(ws + actB * 5);
  __bf16* CTX   = (__bf16*)(ws + actB * 6);
  __bf16* WqT   = (__bf16*)(ws + actB * 7);
  __bf16* WkT   = (__bf16*)(ws + actB * 7 + wB);
  __bf16* WvT   = (__bf16*)(ws + actB * 7 + wB * 2);
  __bf16* WoT   = (__bf16*)(ws + actB * 7 + wB * 3);

  float* out  = (float*)d_out;
  float* mean = out + (size_t)Lq * Bb * DIM;

  const dim3 blk(256);

  // pre-pass: f32 -> bf16 activations; convert+transpose weights
  const int n8 = (Lq * Bb * DIM) / 8;                  // 1,048,576
  cvt_bf16<<<dim3(n8 / 256), blk, 0, stream>>>(q, qb, n8);
  cvt_bf16<<<dim3(n8 / 256), blk, 0, stream>>>(k, kb2, n8);
  cvt_bf16<<<dim3(n8 / 256), blk, 0, stream>>>(v, vb2, n8);
  const dim3 tgrd(DIM / 32, DIM / 32);
  transpose_cvt<<<tgrd, blk, 0, stream>>>(Wq, WqT);
  transpose_cvt<<<tgrd, blk, 0, stream>>>(Wk, WkT);
  transpose_cvt<<<tgrd, blk, 0, stream>>>(Wv, WvT);
  transpose_cvt<<<tgrd, blk, 0, stream>>>(Wout, WoT);

  const dim3 grd((Lq * Bb) / 128, DIM / 128);          // 64 x 8
  gemm_wmma<0><<<grd, blk, 0, stream>>>(qb,  WqT, bq, Qp);
  gemm_wmma<1><<<grd, blk, 0, stream>>>(kb2, WkT, bk, Kp);
  gemm_wmma<2><<<grd, blk, 0, stream>>>(vb2, WvT, bv, VT);

  const size_t smemBytes = (size_t)2 * 32 * 1028 * sizeof(float);   // ~257 KB
  attn_wmma<<<dim3(Lq / 32, Bb), blk, smemBytes, stream>>>(Qp, Kp, VT, mask, CTX, mean);

  gemm_wmma<3><<<grd, blk, 0, stream>>>(CTX, WoT, bout, out);
}
